// EncodeProcessDecode_63144609186418
// MI455X (gfx1250) — matrix operand order repaired and checked by reading, not compiled, not run on hardware
//
#include <hip/hip_runtime.h>

typedef _Float16 half_t;
typedef __attribute__((ext_vector_type(8)))  _Float16 v8h;
typedef __attribute__((ext_vector_type(16))) _Float16 v16h;
typedef __attribute__((ext_vector_type(8)))  float    v8f;

#define LATENT 128

// ---------------- prep kernels ----------------

// W (K x N, row major, f32)  ->  Wt (Np x Kp, row major, f16), zero padded
__global__ void wconv_kernel(const float* __restrict__ W, half_t* __restrict__ Wt,
                             int K, int N, int Kp, int Np) {
  int i = blockIdx.x * 256 + threadIdx.x;
  if (i >= Np * Kp) return;
  int n = i / Kp, k = i - n * Kp;
  float val = (n < N && k < K) ? W[k * N + n] : 0.0f;
  Wt[i] = (half_t)val;
}

// X (R x K f32) -> Xp (R x Kp f16), zero padded columns
__global__ void apad_kernel(const float* __restrict__ X, half_t* __restrict__ Xp,
                            int R, int K, int Kp) {
  int i = blockIdx.x * 256 + threadIdx.x;
  if (i >= R * Kp) return;
  int r = i / Kp, k = i - r * Kp;
  Xp[i] = (half_t)(k < K ? X[r * K + k] : 0.0f);
}

__global__ void zero_kernel(float* __restrict__ p, int n) {
  int i = blockIdx.x * 256 + threadIdx.x;
  if (i < n) p[i] = 0.0f;
}

__global__ void cvt_kernel(const float* __restrict__ s, half_t* __restrict__ d, int n) {
  int i = blockIdx.x * 256 + threadIdx.x;
  if (i < n) d[i] = (half_t)s[i];
}

// scatter-add e_new to agg (by dst), fused with residual e += e_new
__global__ void scatter_kernel(const half_t* __restrict__ e_new, const int* __restrict__ dstI,
                               float* __restrict__ agg, half_t* __restrict__ e, int total) {
  int i = blockIdx.x * 256 + threadIdx.x;
  if (i >= total) return;
  int edge = i >> 7, f = i & 127;
  float val = (float)e_new[i];
  atomicAdd(agg + dstI[edge] * LATENT + f, val);
  e[i] = (half_t)((float)e[i] + val);
}

// ---------------- fused WMMA MLP layer ----------------
// AMODE 0: A = A0 (R x K f16)
// AMODE 1: A = concat[v[dst], v[src], e]   (K = 384), A0 = v, A1 = e
// AMODE 2: A = concat[agg16, v]            (K = 256), A0 = agg16, A1 = v
// Wave tile: 32 rows x 32 cols (2x2 WMMA tiles -> 2 loads per WMMA).
// Block: 8 waves = 64 rows x 128 cols strip; LN (+residual) fused in LDS.

__device__ inline v8h load_msg(const half_t* __restrict__ v, const half_t* __restrict__ e,
                               int gd, int gs, int row, int k) {
  if (k < 128)      return *(const v8h*)(v + gd * LATENT + k);
  else if (k < 256) return *(const v8h*)(v + gs * LATENT + (k - 128));
  else              return *(const v8h*)(e + row * LATENT + (k - 256));
}

__device__ inline v8h load_node(const half_t* __restrict__ agg, const half_t* __restrict__ v,
                                int row, int k) {
  if (k < 128) return *(const v8h*)(agg + row * LATENT + k);
  else         return *(const v8h*)(v + row * LATENT + (k - 128));
}

template <int AMODE, bool RELU, bool LN, bool RES, bool OUTF32>
__global__ __launch_bounds__(256) void mlp_kernel(
    const half_t* __restrict__ A0, const half_t* __restrict__ A1,
    const int* __restrict__ srcI, const int* __restrict__ dstI,
    const half_t* __restrict__ Wt, const float* __restrict__ bias,
    const float* __restrict__ lnG, const float* __restrict__ lnB,
    half_t* __restrict__ outH, const half_t* __restrict__ resH,
    float* __restrict__ outF,
    int R, int K, int nCG, int Nout) {
  const int t    = threadIdx.x & 31;
  const int wave = threadIdx.x >> 5;
  const int rg   = wave >> 2;   // row group (0..1) : 32 rows each
  const int cg   = wave & 3;    // col group (0..3) : 32 cols each
  if constexpr (!LN) { if (cg >= nCG) return; }
  const int hg   = t >> 4;      // half-wave group
  const int lrow = t & 15;
  const int row0 = blockIdx.x * 64;
  const int r0   = row0 + rg * 32;
  int arow0 = r0 + lrow;      if (arow0 >= R) arow0 = R - 1;  // clamp loads
  int arow1 = r0 + 16 + lrow; if (arow1 >= R) arow1 = R - 1;
  const int c0   = cg * 32;
  const int col0 = c0 + lrow, col1 = c0 + 16 + lrow;

  int gd0 = 0, gs0 = 0, gd1 = 0, gs1 = 0;
  if constexpr (AMODE == 1) {
    gd0 = dstI[arow0]; gs0 = srcI[arow0];
    gd1 = dstI[arow1]; gs1 = srcI[arow1];
  }

  v8f acc00 = {}, acc01 = {}, acc10 = {}, acc11 = {};
  const int nk = K >> 5;
  for (int kc = 0; kc < nk; ++kc) {
    const int k0 = kc * 32 + hg * 8;
    v8h alo0, ahi0, alo1, ahi1;
    if constexpr (AMODE == 0) {
      const half_t* ba0 = A0 + arow0 * K;
      const half_t* ba1 = A0 + arow1 * K;
      alo0 = *(const v8h*)(ba0 + k0); ahi0 = *(const v8h*)(ba0 + k0 + 16);
      alo1 = *(const v8h*)(ba1 + k0); ahi1 = *(const v8h*)(ba1 + k0 + 16);
    } else if constexpr (AMODE == 1) {
      alo0 = load_msg(A0, A1, gd0, gs0, arow0, k0);
      ahi0 = load_msg(A0, A1, gd0, gs0, arow0, k0 + 16);
      alo1 = load_msg(A0, A1, gd1, gs1, arow1, k0);
      ahi1 = load_msg(A0, A1, gd1, gs1, arow1, k0 + 16);
    } else {
      alo0 = load_node(A0, A1, arow0, k0); ahi0 = load_node(A0, A1, arow0, k0 + 16);
      alo1 = load_node(A0, A1, arow1, k0); ahi1 = load_node(A0, A1, arow1, k0 + 16);
    }
    const half_t* wb0 = Wt + col0 * K + kc * 32 + hg * 8; // K-halves [8h,8h+8) and [16+8h,24+8h)
    const half_t* wb1 = Wt + col1 * K + kc * 32 + hg * 8;
    v8h blo0 = *(const v8h*)(wb0), bhi0 = *(const v8h*)(wb0 + 16);
    v8h blo1 = *(const v8h*)(wb1), bhi1 = *(const v8h*)(wb1 + 16);
    v16h a0, a1, b0, b1;
#pragma unroll
    for (int i = 0; i < 8; ++i) {
      a0[i] = alo0[i]; a0[i + 8] = ahi0[i];
      a1[i] = alo1[i]; a1[i + 8] = ahi1[i];
      b0[i] = blo0[i]; b0[i + 8] = bhi0[i];
      b1[i] = blo1[i]; b1[i + 8] = bhi1[i];
    }
    acc00 = __builtin_amdgcn_wmma_f32_16x16x32_f16(false, a0, false, b0, (short)0, acc00, false, false);
    acc01 = __builtin_amdgcn_wmma_f32_16x16x32_f16(false, a0, false, b1, (short)0, acc01, false, false);
    acc10 = __builtin_amdgcn_wmma_f32_16x16x32_f16(false, a1, false, b0, (short)0, acc10, false, false);
    acc11 = __builtin_amdgcn_wmma_f32_16x16x32_f16(false, a1, false, b1, (short)0, acc11, false, false);
  }

  float bc0, bc1;
  if constexpr (OUTF32) {
    bc0 = (col0 < Nout) ? bias[col0] : 0.0f;
    bc1 = (col1 < Nout) ? bias[col1] : 0.0f;
  } else {
    bc0 = bias[col0]; bc1 = bias[col1];
  }

  if constexpr (!LN) {
#pragma unroll
    for (int j = 0; j < 8; ++j) {
      int ra = r0 + hg * 8 + j;
      int rb = ra + 16;
      if (ra < R) {
        float y00 = acc00[j] + bc0, y01 = acc01[j] + bc1;
        if constexpr (RELU) { y00 = fmaxf(y00, 0.0f); y01 = fmaxf(y01, 0.0f); }
        if constexpr (OUTF32) {
          if (col0 < Nout) outF[ra * Nout + col0] = y00;
          if (col1 < Nout) outF[ra * Nout + col1] = y01;
        } else {
          outH[ra * LATENT + col0] = (half_t)y00;
          outH[ra * LATENT + col1] = (half_t)y01;
        }
      }
      if (rb < R) {
        float y10 = acc10[j] + bc0, y11 = acc11[j] + bc1;
        if constexpr (RELU) { y10 = fmaxf(y10, 0.0f); y11 = fmaxf(y11, 0.0f); }
        if constexpr (OUTF32) {
          if (col0 < Nout) outF[rb * Nout + col0] = y10;
          if (col1 < Nout) outF[rb * Nout + col1] = y11;
        } else {
          outH[rb * LATENT + col0] = (half_t)y10;
          outH[rb * LATENT + col1] = (half_t)y11;
        }
      }
    }
  } else {
    // 64 x 128 strip in LDS -> per-row LayerNorm (+ optional residual)
    __shared__ float tile_s[64][LATENT + 4];
    __shared__ float red_s[64][4];
    __shared__ float red2_s[64][4];
    __shared__ float mu_s[64], rs_s[64];
#pragma unroll
    for (int j = 0; j < 8; ++j) {
      int lr0 = rg * 32 + hg * 8 + j;
      tile_s[lr0][col0]      = acc00[j] + bc0;
      tile_s[lr0][col1]      = acc01[j] + bc1;
      tile_s[lr0 + 16][col0] = acc10[j] + bc0;
      tile_s[lr0 + 16][col1] = acc11[j] + bc1;
    }
    __syncthreads();
    const int rr = threadIdx.x >> 2;  // 0..63
    const int q  = threadIdx.x & 3;   // quarter of the 128 features
    float s = 0.0f, ss = 0.0f;
#pragma unroll
    for (int i = 0; i < 32; ++i) {
      float xv = tile_s[rr][q * 32 + i];
      s += xv; ss += xv * xv;
    }
    red_s[rr][q] = s; red2_s[rr][q] = ss;
    __syncthreads();
    if (q == 0) {
      float S  = red_s[rr][0] + red_s[rr][1] + red_s[rr][2] + red_s[rr][3];
      float SS = red2_s[rr][0] + red2_s[rr][1] + red2_s[rr][2] + red2_s[rr][3];
      float mu  = S * (1.0f / 128.0f);
      float var = SS * (1.0f / 128.0f) - mu * mu;
      mu_s[rr] = mu;
      rs_s[rr] = rsqrtf(var + 1e-5f);
    }
    __syncthreads();
    int grow = row0 + rr;
    if (grow < R) {
      float mu = mu_s[rr], rstd = rs_s[rr];
#pragma unroll
      for (int i = 0; i < 32; ++i) {
        int cc = q * 32 + i;
        float y = (tile_s[rr][cc] - mu) * rstd * lnG[cc] + lnB[cc];
        if constexpr (RES) y += (float)resH[grow * LATENT + cc];
        outH[grow * LATENT + cc] = (half_t)y;
      }
    }
  }
}

// ---------------- host ----------------

extern "C" void kernel_launch(void* const* d_in, const int* in_sizes, int n_in,
                              void* d_out, int out_size, void* d_ws, size_t ws_size,
                              hipStream_t stream) {
  (void)in_sizes; (void)n_in; (void)out_size; (void)ws_size;
  const int NN = 25000, NE = 100000;
  const float* x    = (const float*)d_in[0];
  const int*   ei   = (const int*)d_in[1];
  const int*   srcI = ei;        // edge_index[0] = j (source)
  const int*   dstI = ei + NE;   // edge_index[1] = i (target)
  const float* ef   = (const float*)d_in[2];
  const float* P[128];
  for (int i = 3; i < 105; ++i) P[i] = (const float*)d_in[i];

  char* ws = (char*)d_ws;
  size_t off = 0;
  auto alloc = [&](size_t bytes) -> void* {
    void* p = ws + off;
    off += (bytes + 255) & ~(size_t)255;
    return p;
  };

  half_t* xp    = (half_t*)alloc((size_t)NN * 32  * sizeof(half_t));
  half_t* efp   = (half_t*)alloc((size_t)NE * 32  * sizeof(half_t));
  half_t* v     = (half_t*)alloc((size_t)NN * 128 * sizeof(half_t));
  half_t* e     = (half_t*)alloc((size_t)NE * 128 * sizeof(half_t));
  half_t* h1    = (half_t*)alloc((size_t)NE * 128 * sizeof(half_t));
  half_t* h2    = (half_t*)alloc((size_t)NE * 128 * sizeof(half_t));
  float*  agg   = (float* )alloc((size_t)NN * 128 * sizeof(float));
  half_t* agg16 = (half_t*)alloc((size_t)NN * 128 * sizeof(half_t));

  // weight conversion: enc_node(3), enc_edge(3), dec(3), 5 steps x (edge 3 + node 3)
  struct WS { int idx, K, N, Kp, Np; };
  WS wl[39]; int nw = 0;
  auto addW = [&](int idx, int K, int N, int Kp, int Np) { wl[nw++] = WS{idx, K, N, Kp, Np}; };
  addW(3, 30, 128, 32, 128);  addW(5, 128, 128, 128, 128);  addW(7, 128, 128, 128, 128);
  addW(11, 3, 128, 32, 128);  addW(13, 128, 128, 128, 128); addW(15, 128, 128, 128, 128);
  addW(19, 128, 128, 128, 128); addW(21, 128, 128, 128, 128); addW(23, 128, 2, 128, 16);
  for (int s = 0; s < 5; ++s) {
    int b = 25 + 16 * s;
    addW(b + 0, 384, 128, 384, 128); addW(b + 2, 128, 128, 128, 128); addW(b + 4, 128, 128, 128, 128);
    addW(b + 8, 256, 128, 256, 128); addW(b + 10, 128, 128, 128, 128); addW(b + 12, 128, 128, 128, 128);
  }
  half_t* Wt[39];
  for (int i = 0; i < nw; ++i) {
    int tot = wl[i].Kp * wl[i].Np;
    Wt[i] = (half_t*)alloc((size_t)tot * sizeof(half_t));
    wconv_kernel<<<(tot + 255) / 256, 256, 0, stream>>>(P[wl[i].idx], Wt[i],
                                                        wl[i].K, wl[i].N, wl[i].Kp, wl[i].Np);
  }

  { int tot = NN * 32; apad_kernel<<<(tot + 255) / 256, 256, 0, stream>>>(x,  xp,  NN, 30, 32); }
  { int tot = NE * 32; apad_kernel<<<(tot + 255) / 256, 256, 0, stream>>>(ef, efp, NE, 3, 32); }

  const int gN = (NN + 63) / 64;  // 64-row blocks
  const int gE = (NE + 63) / 64;

  // ---- encoder: node ----
  mlp_kernel<0, true, false, false, false><<<gN, 256, 0, stream>>>(
      xp, nullptr, nullptr, nullptr, Wt[0], P[4], nullptr, nullptr, h1, nullptr, nullptr, NN, 32, 4, 128);
  mlp_kernel<0, true, false, false, false><<<gN, 256, 0, stream>>>(
      h1, nullptr, nullptr, nullptr, Wt[1], P[6], nullptr, nullptr, h2, nullptr, nullptr, NN, 128, 4, 128);
  mlp_kernel<0, false, true, false, false><<<gN, 256, 0, stream>>>(
      h2, nullptr, nullptr, nullptr, Wt[2], P[8], P[9], P[10], v, nullptr, nullptr, NN, 128, 4, 128);

  // ---- encoder: edge ----
  mlp_kernel<0, true, false, false, false><<<gE, 256, 0, stream>>>(
      efp, nullptr, nullptr, nullptr, Wt[3], P[12], nullptr, nullptr, h1, nullptr, nullptr, NE, 32, 4, 128);
  mlp_kernel<0, true, false, false, false><<<gE, 256, 0, stream>>>(
      h1, nullptr, nullptr, nullptr, Wt[4], P[14], nullptr, nullptr, h2, nullptr, nullptr, NE, 128, 4, 128);
  mlp_kernel<0, false, true, false, false><<<gE, 256, 0, stream>>>(
      h2, nullptr, nullptr, nullptr, Wt[5], P[16], P[17], P[18], e, nullptr, nullptr, NE, 128, 4, 128);

  // ---- processor: 5 InteractionNetwork steps ----
  for (int s = 0; s < 5; ++s) {
    int b  = 25 + 16 * s;
    int wb = 9 + 6 * s;
    // edge MLP (layer0 fuses gather+concat of [v[dst], v[src], e])
    mlp_kernel<1, true, false, false, false><<<gE, 256, 0, stream>>>(
        v, e, srcI, dstI, Wt[wb + 0], P[b + 1], nullptr, nullptr, h1, nullptr, nullptr, NE, 384, 4, 128);
    mlp_kernel<0, true, false, false, false><<<gE, 256, 0, stream>>>(
        h1, nullptr, nullptr, nullptr, Wt[wb + 1], P[b + 3], nullptr, nullptr, h2, nullptr, nullptr, NE, 128, 4, 128);
    mlp_kernel<0, false, true, false, false><<<gE, 256, 0, stream>>>(      // -> e_new in h1
        h2, nullptr, nullptr, nullptr, Wt[wb + 2], P[b + 5], P[b + 6], P[b + 7], h1, nullptr, nullptr, NE, 128, 4, 128);
    // aggregation (scatter-add by dst) fused with e += e_new
    { int tot = NN * 128; zero_kernel<<<(tot + 255) / 256, 256, 0, stream>>>(agg, tot); }
    { int tot = NE * 128; scatter_kernel<<<(tot + 255) / 256, 256, 0, stream>>>(h1, dstI, agg, e, tot); }
    { int tot = NN * 128; cvt_kernel<<<(tot + 255) / 256, 256, 0, stream>>>(agg, agg16, tot); }
    // node MLP (layer0 fuses concat of [agg, v]); final layer fuses LN + residual v += v_new
    mlp_kernel<2, true, false, false, false><<<gN, 256, 0, stream>>>(
        agg16, v, nullptr, nullptr, Wt[wb + 3], P[b + 9], nullptr, nullptr, h2, nullptr, nullptr, NN, 256, 4, 128);
    mlp_kernel<0, true, false, false, false><<<gN, 256, 0, stream>>>(
        h2, nullptr, nullptr, nullptr, Wt[wb + 4], P[b + 11], nullptr, nullptr, h1, nullptr, nullptr, NN, 128, 4, 128);
    mlp_kernel<0, false, true, true, false><<<gN, 256, 0, stream>>>(
        h1, nullptr, nullptr, nullptr, Wt[wb + 5], P[b + 13], P[b + 14], P[b + 15], v, v, nullptr, NN, 128, 4, 128);
  }

  // ---- decoder ----
  mlp_kernel<0, true, false, false, false><<<gN, 256, 0, stream>>>(
      v, nullptr, nullptr, nullptr, Wt[6], P[20], nullptr, nullptr, h1, nullptr, nullptr, NN, 128, 4, 128);
  mlp_kernel<0, true, false, false, false><<<gN, 256, 0, stream>>>(
      h1, nullptr, nullptr, nullptr, Wt[7], P[22], nullptr, nullptr, h2, nullptr, nullptr, NN, 128, 4, 128);
  mlp_kernel<0, false, false, false, true><<<gN, 256, 0, stream>>>(
      h2, nullptr, nullptr, nullptr, Wt[8], P[24], nullptr, nullptr, nullptr, nullptr, (float*)d_out,
      NN, 128, 1, 2);
}
